// VQVAE_62242666053800
// MI455X (gfx1250) — matrix operand order repaired and checked by reading, not compiled, not run on hardware
//
#include <hip/hip_runtime.h>
#include <hip/hip_bf16.h>

// ---------------------------------------------------------------------------
// VQ-VAE forward for MI455X (gfx1250, wave32, WMMA 16x16x32 f16).
// All heavy stages are implicit GEMMs on v_wmma_f32_16x16x32_f16.
// Intermediates are channels-last f16 in workspace; weights pre-swizzled into
// per-lane WMMA B-fragment layout.  Shared 64KB B panels (deconv per-class
// weights, VQ codebook) are staged to LDS with one Tensor-Data-Mover
// descriptor per workgroup (tensor_load_to_lds + s_wait_tensorcnt).
// ---------------------------------------------------------------------------

typedef __attribute__((ext_vector_type(16))) _Float16 v16h;
typedef __attribute__((ext_vector_type(8)))  _Float16 v8h;
typedef __attribute__((ext_vector_type(8)))  float    v8f;
typedef __attribute__((ext_vector_type(4)))  unsigned int u32x4;
typedef __attribute__((ext_vector_type(8)))  unsigned int u32x8;

// Problem constants
#define BATCH 16
#define KCODES 512
#define NPIX  (BATCH*128*128)        // 262144 latent pixels
#define NOUTPIX (BATCH*256*256)      // 1048576 output pixels
#define RECON_ELEMS (BATCH*3*256*256) // 3145728

// Workspace layout (bytes, all 256B aligned)
#define OFF_H      0ull                         // 33554432 halfs
#define OFF_ZE     67108864ull                  // 16777216 floats
#define OFF_QUANT  134217728ull                 // 16777216 halfs
#define OFF_Y      167772160ull                 // 134217728 halfs
#define OFF_IDX    436207616ull                 // 262144 ints
#define OFF_W1B    437256192ull                 // 8192 halfs  (K=64  x N=128 swizzled)
#define OFF_W2B    437272576ull                 // 8192 halfs  (K=128 x N=64)
#define OFF_CBB    437288960ull                 // 32768 halfs (K=64  x N=512)
#define OFF_CNORM  437354496ull                 // 512 floats
#define OFF_WTB    437356544ull                 // 131072 halfs (4 classes x K=256 x N=128)
#define OFF_W3B    437618688ull                 // 2048 halfs  (K=128 x N=16)

__device__ __forceinline__ v16h cat16(v8h lo, v8h hi) {
  return __builtin_shufflevector(lo, hi, 0,1,2,3,4,5,6,7,8,9,10,11,12,13,14,15);
}

__device__ __forceinline__ v8f wmma16(v16h a, v16h b, v8f c) {
  return __builtin_amdgcn_wmma_f32_16x16x32_f16(false, a, false, b, (short)0, c, false, false);
}

// Load A-fragment (16x32 f16) from a dense row-major f16 matrix, row = lane&15.
// ISA layout: lane<16 gets K {0..7,16..23}, lane>=16 gets K {8..15,24..31}.
__device__ __forceinline__ v16h loadA_f16row(const _Float16* row, int kbase, int sel) {
  v8h lo = *(const v8h*)(row + kbase + sel * 8);
  v8h hi = *(const v8h*)(row + kbase + 16 + sel * 8);
  return cat16(lo, hi);
}

// ---------------------------------------------------------------------------
// TDM: DMA a contiguous panel of n_qwords*8 bytes from global to LDS with one
// tensor descriptor (D# group0 + group1, 2D tile, data_size=8B).  Issued by
// one wave; caller must __syncthreads() before other waves consume the LDS.
// ---------------------------------------------------------------------------
__device__ __forceinline__ void tdm_load_panel(void* lds_dst, const void* gsrc,
                                               unsigned n_qwords) {
  unsigned long long ga = (unsigned long long)gsrc;
  unsigned lds_off = (unsigned)(unsigned long long)lds_dst;  // LDS aperture: low 32 = offset
  u32x4 g0;
  g0.x = 1u;                                        // count=1, user mode
  g0.y = lds_off;                                   // lds_addr  (bits 63:32)
  g0.z = (unsigned)ga;                              // global_addr low (bits 95:64)
  g0.w = ((unsigned)(ga >> 32) & 0x01FFFFFFu)       // global_addr hi  (bits 120:96)
         | 0x80000000u;                             // type=2 (bits 127:126)
  u32x8 g1;
  g1[0] = 0x00030000u;                              // wg_mask=0, data_size=3 (8B)
  g1[1] = (n_qwords & 0xFFFFu) << 16;               // tensor_dim0[15:0]  (bits 79:64... 79:48)
  g1[2] = (n_qwords >> 16) | (1u << 16);            // tensor_dim0[31:16], tensor_dim1=1
  g1[3] = (n_qwords & 0xFFFFu) << 16;               // tile_dim0 (bits 127:112)
  g1[4] = 1u;                                       // tile_dim1=1, tile_dim2=0
  g1[5] = n_qwords;                                 // tensor_dim0_stride low 32
  g1[6] = 0u;
  g1[7] = 0u;
  asm volatile("tensor_load_to_lds %0, %1" :: "s"(g0), "s"(g1) : "memory");
  __builtin_amdgcn_s_wait_tensorcnt(0);
}

// ---------------------------------------------------------------------------
// Weight prep: swizzle B matrices into per-lane fragment layout.
// B-frag element j of lane: k = kstep*32 + (j<8 ? (lane>>4)*8 + j : 16 + (lane>>4)*8 + j-8), n = ntile*16 + (lane&15)
// ---------------------------------------------------------------------------
__global__ void k_prep_w1(const float* __restrict__ w1, _Float16* __restrict__ w1B) {
  int id = blockIdx.x * blockDim.x + threadIdx.x;
  if (id >= 8192) return;
  int j = id & 15, lane = (id >> 4) & 31, nt = (id >> 9) & 7, s = id >> 12;
  int sel = lane >> 4, nn = lane & 15;
  int k = s * 32 + ((j < 8) ? (sel * 8 + j) : (16 + sel * 8 + (j - 8)));        // 0..63 (48..63 = zero pad)
  int n = nt * 16 + nn;                 // out channel 0..127
  float val = 0.0f;
  if (k < 48) {
    int c = k >> 4, kh = (k >> 2) & 3, kw = k & 3;
    val = w1[((n * 3 + c) * 4 + kh) * 4 + kw];
  }
  w1B[id] = (_Float16)val;
}

__global__ void k_prep_w2(const float* __restrict__ w2, _Float16* __restrict__ w2B) {
  int id = blockIdx.x * blockDim.x + threadIdx.x;
  if (id >= 8192) return;
  int j = id & 15, lane = (id >> 4) & 31, nt = (id >> 9) & 3, s = id >> 11;
  int sel = lane >> 4, nn = lane & 15;
  int k = s * 32 + ((j < 8) ? (sel * 8 + j) : (16 + sel * 8 + (j - 8)));        // in channel 0..127
  int n = nt * 16 + nn;                 // out dim 0..63
  w2B[id] = (_Float16)w2[n * 128 + k];  // w2 is [64,128,1,1]
}

__global__ void k_prep_cb(const float* __restrict__ cb, _Float16* __restrict__ cbB) {
  int id = blockIdx.x * blockDim.x + threadIdx.x;
  if (id >= 32768) return;
  int j = id & 15, lane = (id >> 4) & 31, nt = (id >> 9) & 31, s = id >> 14;
  int sel = lane >> 4, nn = lane & 15;
  int k = s * 32 + ((j < 8) ? (sel * 8 + j) : (16 + sel * 8 + (j - 8)));        // dim 0..63
  int n = nt * 16 + nn;                 // code 0..511
  cbB[id] = (_Float16)cb[n * 64 + k];
}

__global__ void k_prep_wt(const float* __restrict__ wt, _Float16* __restrict__ wtB) {
  int id = blockIdx.x * blockDim.x + threadIdx.x;
  if (id >= 131072) return;
  int j = id & 15, lane = (id >> 4) & 31, nt = (id >> 9) & 7, s = (id >> 12) & 7;
  int cls = id >> 15;
  int sel = lane >> 4, nn = lane & 15;
  int k = s * 32 + ((j < 8) ? (sel * 8 + j) : (16 + sel * 8 + (j - 8)));        // 0..255 = tap*64 + ci
  int t = k >> 6, ci = k & 63;
  int o = nt * 16 + nn;                 // out channel 0..127
  int rh = cls >> 1, rw = cls & 1;
  int kh = 2 * (t >> 1) + 1 - rh;       // parity-class tap -> kernel index
  int kw = 2 * (t & 1) + 1 - rw;
  // wt layout [I=64][O=128][4][4]
  wtB[id] = (_Float16)wt[((ci * 128 + o) * 4 + kh) * 4 + kw];
}

__global__ void k_prep_w3(const float* __restrict__ w3, _Float16* __restrict__ w3B) {
  int id = blockIdx.x * blockDim.x + threadIdx.x;
  if (id >= 2048) return;
  int j = id & 15, lane = (id >> 4) & 31, s = id >> 9;
  int sel = lane >> 4, nn = lane & 15;
  int k = s * 32 + ((j < 8) ? (sel * 8 + j) : (16 + sel * 8 + (j - 8)));        // in channel 0..127
  float val = (nn < 3) ? w3[nn * 128 + k] : 0.0f;   // pad N 3 -> 16
  w3B[id] = (_Float16)val;
}

__global__ void k_prep_cnorm(const float* __restrict__ cb, float* __restrict__ cnorm,
                             float* __restrict__ loss) {
  int t = blockIdx.x * blockDim.x + threadIdx.x;
  if (t == 0) loss[0] = 0.0f;
  if (t >= KCODES) return;
  float s = 0.0f;
  for (int d = 0; d < 64; ++d) { float v = cb[t * 64 + d]; s += v * v; }
  cnorm[t] = s;
}

// ---------------------------------------------------------------------------
// conv1: 4x4 s2 p1, 3->128, + ReLU.  Implicit GEMM: M=NPIX, K=48(pad 64), N=128.
// ---------------------------------------------------------------------------
__global__ void k_conv1(const float* __restrict__ x, const float* __restrict__ b1,
                        const _Float16* __restrict__ w1B, _Float16* __restrict__ h16) {
  int wid = (blockIdx.x * blockDim.x + threadIdx.x) >> 5;
  if (wid >= NPIX / 16) return;
  int lane = threadIdx.x & 31, r = lane & 15, sel = lane >> 4;
  int mtile = wid;
  int p = mtile * 16 + r;
  int wo = p & 127, ho = (p >> 7) & 127, n = p >> 14;
  v16h a[2];
#pragma unroll
  for (int s = 0; s < 2; ++s) {
#pragma unroll
    for (int j = 0; j < 16; ++j) {
      int koff = (j < 8) ? (sel * 8 + j) : (16 + sel * 8 + (j - 8));
      int k = s * 32 + koff;
      float val = 0.0f;
      if (k < 48) {
        int c = k >> 4, kh = (k >> 2) & 3, kw = k & 3;
        int ih = 2 * ho - 1 + kh, iw = 2 * wo - 1 + kw;
        if ((unsigned)ih < 256u && (unsigned)iw < 256u)
          val = x[((n * 3 + c) * 256 + ih) * 256 + iw];
      }
      a[s][j] = (_Float16)val;
    }
  }
  const v16h* B = (const v16h*)w1B;
#pragma unroll
  for (int nt = 0; nt < 8; ++nt) {
    v8f c = {};
    c = wmma16(a[0], B[(0 * 8 + nt) * 32 + lane], c);
    c = wmma16(a[1], B[(1 * 8 + nt) * 32 + lane], c);
    int o = nt * 16 + (lane & 15);
    float bias = b1[o];
#pragma unroll
    for (int v = 0; v < 8; ++v) {
      int pm = mtile * 16 + v + sel * 8;
      float ov = c[v] + bias;
      ov = ov > 0.0f ? ov : 0.0f;
      h16[pm * 128 + o] = (_Float16)ov;
    }
  }
}

// ---------------------------------------------------------------------------
// conv2: 1x1, 128->64.  GEMM: M=NPIX, K=128, N=64.  Output z_e in f32 (for loss).
// ---------------------------------------------------------------------------
__global__ void k_conv2(const _Float16* __restrict__ h16, const float* __restrict__ b2,
                        const _Float16* __restrict__ w2B, float* __restrict__ ze) {
  int wid = (blockIdx.x * blockDim.x + threadIdx.x) >> 5;
  if (wid >= NPIX / 16) return;
  int lane = threadIdx.x & 31, r = lane & 15, sel = lane >> 4;
  int mtile = wid;
  const _Float16* row = h16 + (mtile * 16 + r) * 128;
  v16h a[4];
#pragma unroll
  for (int s = 0; s < 4; ++s) a[s] = loadA_f16row(row, s * 32, sel);
  const v16h* B = (const v16h*)w2B;
#pragma unroll
  for (int nt = 0; nt < 4; ++nt) {
    v8f c = {};
#pragma unroll
    for (int s = 0; s < 4; ++s) c = wmma16(a[s], B[(s * 4 + nt) * 32 + lane], c);
    int o = nt * 16 + (lane & 15);
    float bias = b2[o];
#pragma unroll
    for (int v = 0; v < 8; ++v) {
      int pm = mtile * 16 + v + sel * 8;
      ze[pm * 64 + o] = c[v] + bias;
    }
  }
}

// ---------------------------------------------------------------------------
// VQ: per flat row argmin_k(||e_k||^2 - 2 f.e_k).  The 64KB swizzled codebook
// panel is TDM-staged to LDS once per block, then 32 WMMA code-tiles + shfl
// reduction per wave.
// ---------------------------------------------------------------------------
__global__ void k_vq(const float* __restrict__ ze, const _Float16* __restrict__ cbB,
                     const float* __restrict__ cnorm, int* __restrict__ idx_i,
                     float* __restrict__ idx_f) {
  __shared__ _Float16 lB[32768];                     // 64KB codebook panel
  if (threadIdx.x < 32)
    tdm_load_panel(lB, cbB, 8192);                   // 8192 qwords = 64KB
  __syncthreads();

  int wid = (blockIdx.x * blockDim.x + threadIdx.x) >> 5;
  int lane = threadIdx.x & 31, r = lane & 15, sel = lane >> 4;
  int mtile = wid;
  const float* zr = ze + (mtile * 16 + r) * 64;
  v16h a[2];
#pragma unroll
  for (int s = 0; s < 2; ++s) {
    v8f c0 = *(const v8f*)(zr + s * 32 + sel * 8);
    v8f c1 = *(const v8f*)(zr + s * 32 + 16 + sel * 8);
#pragma unroll
    for (int j = 0; j < 8; ++j) { a[s][j] = (_Float16)c0[j]; a[s][8 + j] = (_Float16)c1[j]; }
  }
  const v16h* B = (const v16h*)lB;
  float best[8];
  int bcode[8];
#pragma unroll
  for (int v = 0; v < 8; ++v) { best[v] = 3.4e38f; bcode[v] = 0; }
  for (int ct = 0; ct < 32; ++ct) {
    v8f c = {};
    c = wmma16(a[0], B[(0 * 32 + ct) * 32 + lane], c);
    c = wmma16(a[1], B[(1 * 32 + ct) * 32 + lane], c);
    int code = ct * 16 + (lane & 15);
    float cn = cnorm[code];
#pragma unroll
    for (int v = 0; v < 8; ++v) {
      float sv = cn - 2.0f * c[v];
      if (sv < best[v]) { best[v] = sv; bcode[v] = code; }
    }
  }
  // reduce over the 16 lanes holding this row's 16 columns (masks < 16 stay in half)
#pragma unroll
  for (int off = 8; off > 0; off >>= 1) {
#pragma unroll
    for (int v = 0; v < 8; ++v) {
      float ob = __shfl_xor(best[v], off, 32);
      int oc = __shfl_xor(bcode[v], off, 32);
      if (ob < best[v] || (ob == best[v] && oc < bcode[v])) { best[v] = ob; bcode[v] = oc; }
    }
  }
  if ((lane & 15) == 0) {
#pragma unroll
    for (int v = 0; v < 8; ++v) {
      int pm = mtile * 16 + v + sel * 8;
      idx_i[pm] = bcode[v];
      idx_f[pm] = (float)bcode[v];
    }
  }
}

// ---------------------------------------------------------------------------
// Gather quantized rows (f16 for decoder) + commitment loss = 2*mean((q-z)^2).
// ---------------------------------------------------------------------------
__global__ void k_quant_loss(const float* __restrict__ ze, const float* __restrict__ cb,
                             const int* __restrict__ idx_i, _Float16* __restrict__ quant,
                             float* __restrict__ loss) {
  int t = blockIdx.x * blockDim.x + threadIdx.x;
  float ssum = 0.0f;
  if (t < NPIX) {
    int code = idx_i[t];
    const float* e = cb + code * 64;
    const float* z = ze + t * 64;
#pragma unroll 8
    for (int d = 0; d < 64; ++d) {
      float q = e[d];
      quant[t * 64 + d] = (_Float16)q;
      float df = q - z[d];
      ssum += df * df;
    }
  }
#pragma unroll
  for (int off = 16; off > 0; off >>= 1) ssum += __shfl_xor(ssum, off, 32);
  if ((threadIdx.x & 31) == 0)
    atomicAdd(loss, ssum * (2.0f / (float)(NPIX * 64)));
}

// ---------------------------------------------------------------------------
// deconv 4x4 s2 p1, 64->128, + ReLU.  4 output-parity classes, each a 2x2-tap
// conv => implicit GEMM M=NPIX, K=256, N=128 per class.  The class's 64KB B
// panel is TDM-staged to LDS once per block (4 waves share it).
// ---------------------------------------------------------------------------
__global__ void k_deconv(const _Float16* __restrict__ quant, const float* __restrict__ bt,
                         const _Float16* __restrict__ wtB, _Float16* __restrict__ y16) {
  __shared__ _Float16 lB[32768];                     // 64KB class weight panel
  int wid = (blockIdx.x * blockDim.x + threadIdx.x) >> 5;
  int cls = wid >> 14;                 // NPIX/16 = 16384 = 2^14 mtiles per class
  int mtile = wid & 16383;
  if (threadIdx.x < 32)
    tdm_load_panel(lB, wtB + (unsigned long long)cls * 32768ull, 8192);
  __syncthreads();

  int lane = threadIdx.x & 31, r = lane & 15, sel = lane >> 4;
  int rh = cls >> 1, rw = cls & 1;
  int p = mtile * 16 + r;
  int aw = p & 127, ah = (p >> 7) & 127, n = p >> 14;
  v8h zero = {};
  v16h a[8];
#pragma unroll
  for (int s = 0; s < 8; ++s) {
    int t = s >> 1;                    // tap 0..3 (kh_i*2 + kw_i)
    int khi = t >> 1, kwi = t & 1;
    int hp = ah + rh - khi, wp = aw + rw - kwi;
    bool valid = ((unsigned)hp < 128u) && ((unsigned)wp < 128u);
    const _Float16* base = quant + (((n * 128 + hp) * 128) + wp) * 64 + (s & 1) * 32;
    v8h lo = valid ? *(const v8h*)(base + sel * 8) : zero;
    v8h hi = valid ? *(const v8h*)(base + 16 + sel * 8) : zero;
    a[s] = cat16(lo, hi);
  }
  const v16h* B = (const v16h*)lB;
#pragma unroll
  for (int nt = 0; nt < 8; ++nt) {
    v8f c = {};
#pragma unroll
    for (int s = 0; s < 8; ++s) c = wmma16(a[s], B[(s * 8 + nt) * 32 + lane], c);
    int o = nt * 16 + (lane & 15);
    float bias = bt[o];
#pragma unroll
    for (int v = 0; v < 8; ++v) {
      int pm = mtile * 16 + v + sel * 8;
      int aw2 = pm & 127, ah2 = (pm >> 7) & 127, n2 = pm >> 14;
      int oh = 2 * ah2 + rh, ow = 2 * aw2 + rw;
      float val = c[v] + bias;
      val = val > 0.0f ? val : 0.0f;
      y16[(((n2 * 256 + oh) * 256) + ow) * 128 + o] = (_Float16)val;
    }
  }
}

// ---------------------------------------------------------------------------
// conv3: 1x1, 128->3, + sigmoid, output NCHW f32.  GEMM M=NOUTPIX, K=128, N=16(3).
// ---------------------------------------------------------------------------
__global__ void k_conv3(const _Float16* __restrict__ y16, const float* __restrict__ b3,
                        const _Float16* __restrict__ w3B, float* __restrict__ recon) {
  int wid = (blockIdx.x * blockDim.x + threadIdx.x) >> 5;
  if (wid >= NOUTPIX / 16) return;
  int lane = threadIdx.x & 31, r = lane & 15, sel = lane >> 4;
  int mtile = wid;
  const _Float16* row = y16 + (mtile * 16 + r) * 128;
  v16h a[4];
#pragma unroll
  for (int s = 0; s < 4; ++s) a[s] = loadA_f16row(row, s * 32, sel);
  const v16h* B = (const v16h*)w3B;
  v8f c = {};
#pragma unroll
  for (int s = 0; s < 4; ++s) c = wmma16(a[s], B[s * 32 + lane], c);
  int nch = lane & 15;
  if (nch < 3) {
    float bias = b3[nch];
#pragma unroll
    for (int v = 0; v < 8; ++v) {
      int pm = mtile * 16 + v + sel * 8;
      int ow = pm & 255, oh = (pm >> 8) & 255, b = pm >> 16;
      float z = c[v] + bias;
      float sg = 1.0f / (1.0f + expf(-z));
      recon[(b * 3 + nch) * 65536 + oh * 256 + ow] = sg;
    }
  }
}

// ---------------------------------------------------------------------------
extern "C" void kernel_launch(void* const* d_in, const int* in_sizes, int n_in,
                              void* d_out, int out_size, void* d_ws, size_t ws_size,
                              hipStream_t stream) {
  (void)in_sizes; (void)n_in; (void)out_size; (void)ws_size;
  const float* x  = (const float*)d_in[0];
  const float* w1 = (const float*)d_in[1];
  const float* b1 = (const float*)d_in[2];
  const float* w2 = (const float*)d_in[3];
  const float* b2 = (const float*)d_in[4];
  const float* cb = (const float*)d_in[5];
  const float* wt = (const float*)d_in[6];
  const float* bt = (const float*)d_in[7];
  const float* w3 = (const float*)d_in[8];
  const float* b3 = (const float*)d_in[9];

  char* ws = (char*)d_ws;
  _Float16* h16   = (_Float16*)(ws + OFF_H);
  float*    ze    = (float*)   (ws + OFF_ZE);
  _Float16* quant = (_Float16*)(ws + OFF_QUANT);
  _Float16* y16   = (_Float16*)(ws + OFF_Y);
  int*      idx_i = (int*)     (ws + OFF_IDX);
  _Float16* w1B   = (_Float16*)(ws + OFF_W1B);
  _Float16* w2B   = (_Float16*)(ws + OFF_W2B);
  _Float16* cbB   = (_Float16*)(ws + OFF_CBB);
  float*    cnorm = (float*)   (ws + OFF_CNORM);
  _Float16* wtB   = (_Float16*)(ws + OFF_WTB);
  _Float16* w3B   = (_Float16*)(ws + OFF_W3B);

  float* recon = (float*)d_out;
  float* loss  = recon + RECON_ELEMS;
  float* idx_f = recon + RECON_ELEMS + 1;

  // Weight prep (cheap, once per launch; deterministic)
  k_prep_w1<<<32, 256, 0, stream>>>(w1, w1B);
  k_prep_w2<<<32, 256, 0, stream>>>(w2, w2B);
  k_prep_cb<<<128, 256, 0, stream>>>(cb, cbB);
  k_prep_wt<<<512, 256, 0, stream>>>(wt, wtB);
  k_prep_w3<<<8, 256, 0, stream>>>(w3, w3B);
  k_prep_cnorm<<<2, 256, 0, stream>>>(cb, cnorm, loss);

  // Pipeline (stream-ordered)
  k_conv1<<<4096, 128, 0, stream>>>(x, b1, w1B, h16);            // 16384 waves
  k_conv2<<<4096, 128, 0, stream>>>(h16, b2, w2B, ze);           // 16384 waves
  k_vq<<<4096, 128, 0, stream>>>(ze, cbB, cnorm, idx_i, idx_f);  // 16384 waves
  k_quant_loss<<<1024, 256, 0, stream>>>(ze, cb, idx_i, quant, loss);
  k_deconv<<<16384, 128, 0, stream>>>(quant, bt, wtB, y16);      // 65536 waves
  k_conv3<<<16384, 128, 0, stream>>>(y16, b3, w3B, recon);       // 65536 waves
}
